// MobiusGRU_c_38190849196687
// MI455X (gfx1250) — matrix operand order repaired and checked by reading, not compiled, not run on hardware
//
#include <hip/hip_runtime.h>
#include <hip/hip_bf16.h>
#include <math.h>

// ---------------------------------------------------------------------------
// Möbius (hyperbolic) GRU for MI455X / gfx1250.  B=64, S=512, D=H=512.
//
// Phase 1 (parallel):  xh = expmap0(x);  Ux = mmatvec(W_i*, xh) for all (b,s)
//   via one 32768 x 1536 x 512 bf16 WMMA GEMM + fused row scaling.
// Phase 2 (one persistent kernel): the scan is independent per batch row, so
//   4 workgroups (16 rows each = one WMMA M-tile) run all 512 steps with only
//   workgroup barriers.  Per step: WMMA GEMM (h @ [W_hr;W_hz]^T, A from LDS,
//   weights streamed from L2) -> per-row gate math (1 wave = 1 row, all dots
//   are wave-shuffle reductions) -> WMMA GEMM (rh @ W_hhh^T) -> Mobius update.
// ---------------------------------------------------------------------------

#define BB 64
#define SS 512
#define DD 512
#define HH 512
#define HSTRIDE 520   // padded bf16 row stride: ds_load_b128 conflict-free

typedef __attribute__((ext_vector_type(8)))  float  v8f;
typedef __attribute__((ext_vector_type(8)))  __bf16 v8bf;
typedef __attribute__((ext_vector_type(16))) __bf16 v16bf;

// ------------------------------- math helpers ------------------------------

__device__ __forceinline__ float dev_softplus_neg(float kv) {
  float z = -kv;
  return (z > 20.0f) ? z : log1pf(expf(z));
}
__device__ __forceinline__ float dev_tan_k(float v, float sk) {
  float a = fminf(fmaxf(v * sk, -15.0f), 15.0f);
  return tanhf(a) / sk;
}
__device__ __forceinline__ float dev_artan_k(float v, float sk) {
  float a = fminf(fmaxf(v * sk, -1.0f + 1e-5f), 1.0f - 1e-5f);
  return atanhf(a) / sk;
}

__device__ __forceinline__ float wave_sum(float v) {
#pragma unroll
  for (int o = 16; o > 0; o >>= 1) v += __shfl_down(v, o, 32);
  return __shfl(v, 0, 32);
}

// Block reduction for the 256-thread setup kernels.
__device__ __forceinline__ float blk_sum256(float v, float* s) {
#pragma unroll
  for (int o = 16; o > 0; o >>= 1) v += __shfl_down(v, o, 32);
  const int lane = threadIdx.x & 31, w = threadIdx.x >> 5;
  if (lane == 0) s[w] = v;
  __syncthreads();
  if (w == 0) {
    float r = (lane < 8) ? s[lane] : 0.0f;
#pragma unroll
    for (int o = 4; o > 0; o >>= 1) r += __shfl_down(r, o, 32);
    if (lane == 0) s[0] = r;
  }
  __syncthreads();
  float out = s[0];
  __syncthreads();
  return out;
}

// ------------------------------ setup kernels ------------------------------

__global__ __launch_bounds__(256) void k_cvt_bf16(const float* __restrict__ src,
                                                  __bf16* __restrict__ dst, int n) {
  int i = blockIdx.x * 256 + threadIdx.x;
  if (i < n) dst[i] = (__bf16)src[i];
}

// xh = expmap0(x): one workgroup per row of 512; emits bf16 row + its norm.
__global__ __launch_bounds__(256) void k_expmap(const float* __restrict__ x,
                                                const float* __restrict__ kp,
                                                __bf16* __restrict__ xh_bf,
                                                float* __restrict__ vnx) {
  __shared__ float sred[9];
  const int row = blockIdx.x;
  const float* xr = x + (size_t)row * DD;
  float v0 = xr[threadIdx.x], v1 = xr[threadIdx.x + 256];
  float ss = blk_sum256(v0 * v0 + v1 * v1, sred);
  float sp = dev_softplus_neg(kp[0]);
  float sk = sqrtf(fmaxf(sp, 1e-30f));
  float un = fmaxf(sqrtf(ss), 1e-15f);
  float t  = dev_tan_k(un, sk);
  float sc = t / un;
  xh_bf[(size_t)row * DD + threadIdx.x]       = (__bf16)(v0 * sc);
  xh_bf[(size_t)row * DD + threadIdx.x + 256] = (__bf16)(v1 * sc);
  if (threadIdx.x == 0) vnx[row] = t;
}

// C[M,N] = A[M,K](bf16) @ W[N,K](bf16)^T, f32 acc; 4 waves, 16x16 tile each.
__global__ __launch_bounds__(128) void k_gemm_bf16(
    const __bf16* __restrict__ A, int lda, const __bf16* __restrict__ W, int ldw,
    float* __restrict__ C, int ldc, int K) {
  const int wave = threadIdx.x >> 5, lane = threadIdx.x & 31;
  const int hi = lane >> 4, ln = lane & 15;
  const int tileM = blockIdx.y * 16;
  const int n = blockIdx.x * 64 + wave * 16 + ln;
  const int m = tileM + ln;
  const int kb = hi ? 8 : 0, kh = hi ? 16 : 0;
  const __bf16* __restrict__ arow = A + (size_t)m * lda;
  const __bf16* __restrict__ wrow = W + (size_t)n * ldw;
  v8f acc = {};
  for (int k0 = 0; k0 < K; k0 += 32) {
    __builtin_prefetch(wrow + k0 + 64, 0, 1);
    v8bf alo = *(const v8bf*)(arow + k0 + kb);
    v8bf ahi = *(const v8bf*)(arow + k0 + 16 + kb);
    v16bf av, bv;
    v8bf blo = *(const v8bf*)(wrow + k0 + kb);
    v8bf bhi = *(const v8bf*)(wrow + k0 + 16 + kb);
#pragma unroll
    for (int i = 0; i < 8; ++i) { av[i] = alo[i]; av[i + 8] = ahi[i]; bv[i] = blo[i]; bv[i + 8] = bhi[i]; }
    acc = __builtin_amdgcn_wmma_f32_16x16x32_bf16(false, av, false, bv,
                                                  (short)0, acc, false, false);
  }
  const int rbase = tileM + (hi ? 8 : 0);
#pragma unroll
  for (int i = 0; i < 8; ++i) C[(size_t)(rbase + i) * ldc + n] = acc[i];
}

// fused mmatvec tail on Ux: scale each (row, gate) block of 512, record norm.
__global__ __launch_bounds__(256) void k_uxscale(const float* __restrict__ kp,
                                                 float* __restrict__ Ux,
                                                 const float* __restrict__ vnx,
                                                 float* __restrict__ uxn) {
  __shared__ float sred[9];
  const int g = blockIdx.x, row = blockIdx.y;
  float* mv = Ux + (size_t)row * (3 * HH) + (size_t)g * HH;
  float v0 = mv[threadIdx.x], v1 = mv[threadIdx.x + 256];
  float ss  = blk_sum256(v0 * v0 + v1 * v1, sred);
  float sp  = dev_softplus_neg(kp[0]);
  float sk  = sqrtf(fmaxf(sp, 1e-30f));
  float vn  = fmaxf(vnx[row], 1e-15f);
  float mvn = fmaxf(sqrtf(ss), 1e-15f);
  float t   = dev_tan_k(mvn / vn * dev_artan_k(vn, sk), sk);
  float sc  = t / mvn;
  mv[threadIdx.x]       = v0 * sc;
  mv[threadIdx.x + 256] = v1 * sc;
  if (threadIdx.x == 0) uxn[(size_t)row * 3 + g] = t;
}

// --------------------- persistent fused scan kernel ------------------------
// grid = 4 WGs x 512 threads (16 waves).  Wave w owns batch row blockIdx.x*16+w.
// LDS: h_bf[16][HSTRIDE] bf16, rh_bf[16][HSTRIDE] bf16, mv[16][1024] f32.

#define SM_HBF   0
#define SM_RHB   (16 * HSTRIDE * 2)
#define SM_MV    (2 * 16 * HSTRIDE * 2)
#define SM_BYTES (SM_MV + 16 * 1024 * 4)

__global__ __launch_bounds__(512) void k_scan(
    const float* __restrict__ kp, const __bf16* __restrict__ Whh_bf,
    const float* __restrict__ Ux, const float* __restrict__ uxn,
    const float* __restrict__ bias, float* __restrict__ out) {
  extern __shared__ char smem[];
  __bf16* h_bf = (__bf16*)(smem + SM_HBF);
  __bf16* rhb  = (__bf16*)(smem + SM_RHB);
  float*  mv   = (float*)(smem + SM_MV);

  const int tid  = threadIdx.x;
  const int w    = tid >> 5;          // wave id == owned row (0..15)
  const int lane = tid & 31;
  const int hi   = lane >> 4, ln = lane & 15;
  const int kb   = hi ? 8 : 0, kh = hi ? 16 : 0, rb = hi ? 8 : 0;
  const int b    = blockIdx.x * 16 + w;

  const float sp   = dev_softplus_neg(kp[0]);
  const float kneg = -sp;
  const float sk   = sqrtf(fmaxf(sp, 1e-30f));

  float hreg[16];
#pragma unroll
  for (int j = 0; j < 16; ++j) hreg[j] = 0.0f;
  float hn2 = 0.0f;
  for (int i = tid; i < 16 * HSTRIDE; i += 512) h_bf[i] = (__bf16)0.0f;
  __syncthreads();

#pragma unroll 1
  for (int t = 0; t < SS; ++t) {
    const size_t uxrow = ((size_t)b * SS + t) * (3 * HH);
    __builtin_prefetch(Ux + uxrow + lane * 48, 0, 1);   // warm this step's Ux

    // -------- G1: mv[16][1024] = h @ [W_hr ; W_hz]^T (WMMA, A from LDS) ----
#pragma unroll 1
    for (int tt = 4 * w; tt < 4 * w + 4; ++tt) {
      const int nlog  = tt * 16 + ln;
      const int nphys = (nlog < 512) ? nlog : nlog + 512;  // z -> Whh rows 1024+
      const __bf16* __restrict__ wr = Whh_bf + (size_t)nphys * HH;
      v8f acc = {};
      for (int k0 = 0; k0 < HH; k0 += 32) {
        v8bf alo = *(const v8bf*)(h_bf + ln * HSTRIDE + k0 + kb);
        v8bf ahi = *(const v8bf*)(h_bf + ln * HSTRIDE + k0 + 16 + kb);
        v16bf av, bv;
        v8bf blo = *(const v8bf*)(wr + k0 + kb);
        v8bf bhi = *(const v8bf*)(wr + k0 + 16 + kb);
#pragma unroll
        for (int i = 0; i < 8; ++i) { av[i] = alo[i]; av[i + 8] = ahi[i]; bv[i] = blo[i]; bv[i + 8] = bhi[i]; }
        acc = __builtin_amdgcn_wmma_f32_16x16x32_bf16(false, av, false, bv,
                                                      (short)0, acc, false, false);
      }
#pragma unroll
      for (int i = 0; i < 8; ++i) mv[(rb + i) * 1024 + nlog] = acc[i];
    }
    __syncthreads();

    // -------- P2: gates r,z + rh (wave w handles row b; wave reductions) ---
    const float vn  = fmaxf(sqrtf(hn2), 1e-15f);
    const float avn = dev_artan_k(vn, sk);
    float rreg[16], zreg[16];
#pragma unroll 1
    for (int g = 0; g < 2; ++g) {
      const int gI = g ? 2 : 0;                 // W_ir/b_r or W_iz/b_z
      float mvv[16]; float s2 = 0.0f;
#pragma unroll
      for (int j = 0; j < 16; ++j) {
        mvv[j] = mv[w * 1024 + g * HH + lane + 32 * j];
        s2 += mvv[j] * mvv[j];
      }
      float mvn = fmaxf(sqrtf(wave_sum(s2)), 1e-15f);
      float tA  = dev_tan_k(mvn / vn * avn, sk);
      float sA  = tA / mvn;
      float uxv[16]; float sab = 0.0f;
#pragma unroll
      for (int j = 0; j < 16; ++j) {
        uxv[j] = Ux[uxrow + (size_t)gI * HH + lane + 32 * j];
        sab += (sA * mvv[j]) * uxv[j];
      }
      float ab = wave_sum(sab);
      float bn = uxn[((size_t)b * SS + t) * 3 + gI];
      float a2 = tA * tA, b2 = bn * bn;
      float ca  = 1.0f - 2.0f * kneg * ab - kneg * b2;
      float cb  = 1.0f + kneg * a2;
      float den = fmaxf(1.0f - 2.0f * kneg * ab + kneg * kneg * a2 * b2, 1e-15f);
      float cvec[16]; float sc2 = 0.0f, sbb = 0.0f, scb = 0.0f;
#pragma unroll
      for (int j = 0; j < 16; ++j) {
        cvec[j] = (ca * (sA * mvv[j]) + cb * uxv[j]) / den;
        float bbv = bias[gI * HH + lane + 32 * j];
        uxv[j] = bbv;                         // reuse registers for bias
        sc2 += cvec[j] * cvec[j];
        sbb += bbv * bbv;
        scb += cvec[j] * bbv;
      }
      float c2 = wave_sum(sc2), bb2 = wave_sum(sbb), cab = wave_sum(scb);
      float ca2  = 1.0f - 2.0f * kneg * cab - kneg * bb2;
      float cb2  = 1.0f + kneg * c2;
      float den2 = fmaxf(1.0f - 2.0f * kneg * cab + kneg * kneg * c2 * bb2, 1e-15f);
      float sd2 = 0.0f;
#pragma unroll
      for (int j = 0; j < 16; ++j) {
        cvec[j] = (ca2 * cvec[j] + cb2 * uxv[j]) / den2;
        sd2 += cvec[j] * cvec[j];
      }
      float dn  = fmaxf(sqrtf(wave_sum(sd2)), 1e-15f);
      float lsc = dev_artan_k(dn, sk) / dn;
#pragma unroll
      for (int j = 0; j < 16; ++j) {
        float gate = 1.0f / (1.0f + expf(-lsc * cvec[j]));
        if (g == 0) rreg[j] = gate; else zreg[j] = gate;
      }
    }
    // rh = mpmul(r, h)
    float wv[16]; float srw = 0.0f;
#pragma unroll
    for (int j = 0; j < 16; ++j) { wv[j] = rreg[j] * hreg[j]; srw += wv[j] * wv[j]; }
    float wvn = fmaxf(sqrtf(wave_sum(srw)), 1e-15f);
    float trh = dev_tan_k(wvn / vn * avn, sk);   // == ||rh||
    float src = trh / wvn;
#pragma unroll
    for (int j = 0; j < 16; ++j)
      rhb[w * HSTRIDE + lane + 32 * j] = (__bf16)(src * wv[j]);
    __syncthreads();

    // -------- G3: mv[16][512] = rh @ W_hhh^T (WMMA) ------------------------
#pragma unroll 1
    for (int tt = 2 * w; tt < 2 * w + 2; ++tt) {
      const int n = tt * 16 + ln;
      const __bf16* __restrict__ wr = Whh_bf + (size_t)(HH + n) * HH;
      v8f acc = {};
      for (int k0 = 0; k0 < HH; k0 += 32) {
        v8bf alo = *(const v8bf*)(rhb + ln * HSTRIDE + k0 + kb);
        v8bf ahi = *(const v8bf*)(rhb + ln * HSTRIDE + k0 + 16 + kb);
        v16bf av, bv;
        v8bf blo = *(const v8bf*)(wr + k0 + kb);
        v8bf bhi = *(const v8bf*)(wr + k0 + 16 + kb);
#pragma unroll
        for (int i = 0; i < 8; ++i) { av[i] = alo[i]; av[i + 8] = ahi[i]; bv[i] = blo[i]; bv[i + 8] = bhi[i]; }
        acc = __builtin_amdgcn_wmma_f32_16x16x32_bf16(false, av, false, bv,
                                                      (short)0, acc, false, false);
      }
#pragma unroll
      for (int i = 0; i < 8; ++i) mv[(rb + i) * HH + n] = acc[i];
    }
    __syncthreads();

    // -------- P4: h_tilde, delta, hn; write outputs ------------------------
    {
      float vnr = fmaxf(trh, 1e-15f);
      float mvv[16]; float s2 = 0.0f;
#pragma unroll
      for (int j = 0; j < 16; ++j) {
        mvv[j] = mv[w * HH + lane + 32 * j];
        s2 += mvv[j] * mvv[j];
      }
      float mvn = fmaxf(sqrtf(wave_sum(s2)), 1e-15f);
      float tA  = dev_tan_k(mvn / vnr * dev_artan_k(vnr, sk), sk);
      float sA  = tA / mvn;
      float uxv[16]; float sab = 0.0f;
#pragma unroll
      for (int j = 0; j < 16; ++j) {
        uxv[j] = Ux[uxrow + (size_t)HH + lane + 32 * j];    // gate 1 = W_ihh
        sab += (sA * mvv[j]) * uxv[j];
      }
      float ab = wave_sum(sab);
      float bn = uxn[((size_t)b * SS + t) * 3 + 1];
      float a2 = tA * tA, b2 = bn * bn;
      float ca  = 1.0f - 2.0f * kneg * ab - kneg * b2;
      float cb  = 1.0f + kneg * a2;
      float den = fmaxf(1.0f - 2.0f * kneg * ab + kneg * kneg * a2 * b2, 1e-15f);
      float cvec[16]; float sc2 = 0.0f, sbb = 0.0f, scb = 0.0f;
#pragma unroll
      for (int j = 0; j < 16; ++j) {
        cvec[j] = (ca * (sA * mvv[j]) + cb * uxv[j]) / den;
        float bbv = bias[1 * HH + lane + 32 * j];           // b_h
        uxv[j] = bbv;
        sc2 += cvec[j] * cvec[j];
        sbb += bbv * bbv;
        scb += cvec[j] * bbv;
      }
      float c2 = wave_sum(sc2), bb2 = wave_sum(sbb), cab = wave_sum(scb);
      float ca2  = 1.0f - 2.0f * kneg * cab - kneg * bb2;
      float cb2  = 1.0f + kneg * c2;
      float den2 = fmaxf(1.0f - 2.0f * kneg * cab + kneg * kneg * c2 * bb2, 1e-15f);
      float ht[16]; float sht = 0.0f, shht = 0.0f;
#pragma unroll
      for (int j = 0; j < 16; ++j) {
        ht[j] = (ca2 * cvec[j] + cb2 * uxv[j]) / den2;      // h_tilde
        sht  += ht[j] * ht[j];
        shht += hreg[j] * ht[j];
      }
      float ht2 = wave_sum(sht), hht = wave_sum(shht);
      // delta = madd(-h, h_tilde): a2=hn2, b2=ht2, ab=-hht
      float ca3  = 1.0f + 2.0f * kneg * hht - kneg * ht2;
      float cb3  = 1.0f + kneg * hn2;
      float den3 = fmaxf(1.0f + 2.0f * kneg * hht + kneg * kneg * hn2 * ht2, 1e-15f);
      float delta[16]; float sdd = 0.0f;
#pragma unroll
      for (int j = 0; j < 16; ++j) {
        delta[j] = (ca3 * (-hreg[j]) + cb3 * ht[j]) / den3;
        sdd += delta[j] * delta[j];
      }
      float dn = fmaxf(sqrtf(wave_sum(sdd)), 1e-15f);
      float zd[16]; float swv = 0.0f;
#pragma unroll
      for (int j = 0; j < 16; ++j) { zd[j] = zreg[j] * delta[j]; swv += zd[j] * zd[j]; }
      float wvn2 = fmaxf(sqrtf(wave_sum(swv)), 1e-15f);
      float t2   = dev_tan_k(wvn2 / dn * dev_artan_k(dn, sk), sk);
      float szd  = t2 / wvn2;
      float shzd = 0.0f;
#pragma unroll
      for (int j = 0; j < 16; ++j) { zd[j] *= szd; shzd += hreg[j] * zd[j]; }
      float hzd = wave_sum(shzd);
      // hn = madd(h, zd): a2=hn2, b2=t2^2, ab=hzd
      float b2n  = t2 * t2;
      float ca4  = 1.0f - 2.0f * kneg * hzd - kneg * b2n;
      float cb4  = 1.0f + kneg * hn2;
      float den4 = fmaxf(1.0f - 2.0f * kneg * hzd + kneg * kneg * hn2 * b2n, 1e-15f);
      float snn = 0.0f;
#pragma unroll
      for (int j = 0; j < 16; ++j) {
        float hnv = (ca4 * hreg[j] + cb4 * zd[j]) / den4;
        out[(size_t)t * (BB * HH) + (size_t)b * HH + lane + 32 * j] = hnv;
        if (t == SS - 1)
          out[(size_t)SS * (BB * HH) + (size_t)b * HH + lane + 32 * j] = hnv;
        hreg[j] = hnv;
        snn += hnv * hnv;
        h_bf[w * HSTRIDE + lane + 32 * j] = (__bf16)hnv;
      }
      hn2 = wave_sum(snn);
    }
    __syncthreads();
  }
}

// ------------------------------- host driver -------------------------------

extern "C" void kernel_launch(void* const* d_in, const int* in_sizes, int n_in,
                              void* d_out, int out_size, void* d_ws, size_t ws_size,
                              hipStream_t stream) {
  (void)in_sizes; (void)n_in; (void)out_size; (void)ws_size;
  const float* x    = (const float*)d_in[0];
  const float* kp   = (const float*)d_in[1];
  const float* Wih  = (const float*)d_in[2];
  const float* Whh  = (const float*)d_in[3];
  const float* bias = (const float*)d_in[4];
  float* out = (float*)d_out;

  char*  ws  = (char*)d_ws;
  size_t off = 0;
  auto alloc = [&](size_t bytes) -> char* {
    char* p = ws + off;
    off = (off + bytes + 255) & ~(size_t)255;
    return p;
  };
  const size_t NROW = (size_t)BB * SS;
  __bf16* Wih_bf = (__bf16*)alloc((size_t)3 * HH * DD * 2);
  __bf16* Whh_bf = (__bf16*)alloc((size_t)3 * HH * HH * 2);
  __bf16* xh_bf  = (__bf16*)alloc(NROW * DD * 2);
  float*  vnx    = (float*)alloc(NROW * 4);
  float*  Ux     = (float*)alloc(NROW * 3 * HH * 4);
  float*  uxn    = (float*)alloc(NROW * 3 * 4);

  // weights -> bf16
  {
    int n = 3 * HH * DD;
    k_cvt_bf16<<<(n + 255) / 256, 256, 0, stream>>>(Wih, Wih_bf, n);
    k_cvt_bf16<<<(n + 255) / 256, 256, 0, stream>>>(Whh, Whh_bf, n);
  }
  // xh = expmap0(x)
  k_expmap<<<(int)NROW, 256, 0, stream>>>(x, kp, xh_bf, vnx);
  // Ux raw = xh @ Wih^T  (32768 x 1536 x 512, WMMA)
  k_gemm_bf16<<<dim3((3 * HH) / 64, (int)(NROW / 16)), 128, 0, stream>>>(
      xh_bf, DD, Wih_bf, DD, Ux, 3 * HH, DD);
  // fused mmatvec tail
  k_uxscale<<<dim3(3, (int)NROW), 256, 0, stream>>>(kp, Ux, vnx, uxn);

  // persistent fused scan: 4 WGs, 16 rows each, zero cross-WG sync
  (void)hipFuncSetAttribute((const void*)k_scan,
                            hipFuncAttributeMaxDynamicSharedMemorySize, SM_BYTES);
  k_scan<<<BB / 16, 512, SM_BYTES, stream>>>(kp, Whh_bf, Ux, uxn, bias, out);
}
